// AdaptiveProjection_82566451299001
// MI455X (gfx1250) — matrix operand order repaired and checked by reading, not compiled, not run on hardware
//
#include <hip/hip_runtime.h>

// Problem constants (from reference): B=4, S=4096, D=1024, O=1024, E=4
#define BTOK 4
#define SEQ  4096
#define DIM  1024
#define OUTD 1024
#define NEXP 4
#define MTOT (BTOK * SEQ)   // 16384 tokens

typedef __attribute__((ext_vector_type(16))) __bf16 v16bf;
typedef __attribute__((ext_vector_type(8)))  float  v8f;

union FragU {
    unsigned int u[8];
    v16bf v;
};

// fp32 -> bf16 round-to-nearest-even
__device__ __forceinline__ unsigned short f2bf(float f) {
    unsigned int u = __float_as_uint(f);
    u += 0x7FFFu + ((u >> 16) & 1u);
    return (unsigned short)(u >> 16);
}

__device__ __forceinline__ uint2 pack4bf(float4 v) {
    uint2 pk;
    pk.x = (unsigned)f2bf(v.x) | ((unsigned)f2bf(v.y) << 16);
    pk.y = (unsigned)f2bf(v.z) | ((unsigned)f2bf(v.w) << 16);
    return pk;
}

// Low 32 bits of a generic pointer to __shared__ data == LDS byte offset
// (AS3 -> generic addrspacecast places the LDS offset in the low dword).
__device__ __forceinline__ unsigned lds_off(const void* p) {
    return (unsigned)(uintptr_t)p;
}

// CDNA5 async DMA: global -> LDS, 16 bytes per lane, tracked by ASYNCcnt.
// GVS mode: addr = SADDR(64b, uniform) + VADDR[31:0] (per-lane byte offset).
__device__ __forceinline__ void async_ld16(unsigned ldsDst, const void* base,
                                           unsigned gOff) {
    asm volatile("global_load_async_to_lds_b128 %0, %1, %2 offset:0"
                 :
                 : "v"(ldsDst), "v"(gOff), "s"(base)
                 : "memory");
}

__device__ __forceinline__ void wait_async0() {
    asm volatile("s_wait_asynccnt 0x0" ::: "memory");
}

// ---------------------------------------------------------------------------
// Kernel 1: per-token gate (softmax over 4 experts) + convert x -> bf16.
// One wave32 per token; 8 waves per 256-thread block.
// ---------------------------------------------------------------------------
__global__ __launch_bounds__(256) void gate_and_cvt_x(
    const float* __restrict__ x,      // [MTOT, DIM]
    const float* __restrict__ gw,     // [NEXP, DIM]
    const float* __restrict__ gb,     // [NEXP]
    float* __restrict__ gate,         // [MTOT, NEXP]
    unsigned short* __restrict__ xb)  // [MTOT, DIM] bf16
{
    const int wave = threadIdx.x >> 5;
    const int lane = threadIdx.x & 31;
    const int n = blockIdx.x * 8 + wave;

    const float* xr = x + (size_t)n * DIM;
    unsigned short* xbr = xb + (size_t)n * DIM;

    float a0 = 0.f, a1 = 0.f, a2 = 0.f, a3 = 0.f;
#pragma unroll
    for (int it = 0; it < DIM / 128; ++it) {
        const int j = it * 128 + lane * 4;
        float4 xv = *(const float4*)(xr + j);
        *(uint2*)(xbr + j) = pack4bf(xv);
        float4 w0 = *(const float4*)(gw + 0 * DIM + j);
        float4 w1 = *(const float4*)(gw + 1 * DIM + j);
        float4 w2 = *(const float4*)(gw + 2 * DIM + j);
        float4 w3 = *(const float4*)(gw + 3 * DIM + j);
        a0 += xv.x * w0.x + xv.y * w0.y + xv.z * w0.z + xv.w * w0.w;
        a1 += xv.x * w1.x + xv.y * w1.y + xv.z * w1.z + xv.w * w1.w;
        a2 += xv.x * w2.x + xv.y * w2.y + xv.z * w2.z + xv.w * w2.w;
        a3 += xv.x * w3.x + xv.y * w3.y + xv.z * w3.z + xv.w * w3.w;
    }
    // wave32 butterfly reduction (all lanes end with the full sums)
#pragma unroll
    for (int s = 16; s > 0; s >>= 1) {
        a0 += __shfl_xor(a0, s, 32);
        a1 += __shfl_xor(a1, s, 32);
        a2 += __shfl_xor(a2, s, 32);
        a3 += __shfl_xor(a3, s, 32);
    }
    a0 += gb[0]; a1 += gb[1]; a2 += gb[2]; a3 += gb[3];
    const float m  = fmaxf(fmaxf(a0, a1), fmaxf(a2, a3));
    const float e0 = __expf(a0 - m);
    const float e1 = __expf(a1 - m);
    const float e2 = __expf(a2 - m);
    const float e3 = __expf(a3 - m);
    const float inv = 1.f / (e0 + e1 + e2 + e3);
    if (lane == 0) {
        float4 g = make_float4(e0 * inv, e1 * inv, e2 * inv, e3 * inv);
        *(float4*)(gate + (size_t)n * 4) = g;
    }
}

// ---------------------------------------------------------------------------
// Kernel 2: W_experts fp32 [E,O,D] -> bf16 (same layout).
// ---------------------------------------------------------------------------
__global__ __launch_bounds__(256) void cvt_w(
    const float* __restrict__ w, unsigned short* __restrict__ wb)
{
    const size_t i = ((size_t)blockIdx.x * 256 + threadIdx.x) * 4;
    float4 v = *(const float4*)(w + i);
    *(uint2*)(wb + i) = pack4bf(v);
}

// ---------------------------------------------------------------------------
// Kernel 3: fused MoE GEMM.
// out[m, o] = sum_e gate[m,e] * sum_d xb[m,d] * wb[e,o,d]
// 256 threads (8 waves), 128(M) x 64(N) tile, K-step 32, double-buffered LDS
// filled by GLOBAL_LOAD_ASYNC_TO_LDS_B128 (ASYNCcnt), one barrier per K-step.
// Waves: 4 (m) x 2 (n); each wave owns 2x2 16x16 C tiles x 4 expert accums.
// ---------------------------------------------------------------------------
__global__ __launch_bounds__(256) void moe_gemm(
    const unsigned short* __restrict__ xb,   // [MTOT, DIM] bf16
    const unsigned short* __restrict__ wb,   // [NEXP, OUTD, DIM] bf16
    const float* __restrict__ gate,          // [MTOT, NEXP]
    float* __restrict__ out)                 // [MTOT, OUTD]
{
    __shared__ __align__(16) unsigned short sA[2][128 * 32];        // 2 x  8 KB
    __shared__ __align__(16) unsigned short sB[2][NEXP * 64 * 32];  // 2 x 16 KB

    const int tid    = threadIdx.x;
    const int lane   = tid & 31;
    const int waveId = tid >> 5;
    const int wm     = waveId & 3;   // 0..3 -> m offset wm*32
    const int wn     = waveId >> 2;  // 0..1 -> n offset wn*32
    const int mBase  = blockIdx.y * 128;
    const int nBase  = blockIdx.x * 64;

    // --- async staging of one K-step tile into LDS buffer `b` ---
    auto stage = [&](int b, int kk) {
        // A tile: 128 rows x 32 bf16 = 512 x 16B chunks, 2 per thread
#pragma unroll
        for (int c = 0; c < 2; ++c) {
            const int idx = tid + c * 256;
            const int row = idx >> 2;
            const int cc  = idx & 3;
            const unsigned l = lds_off(&sA[b][row * 32 + cc * 8]);
            const unsigned g =
                (unsigned)(((size_t)(mBase + row) * DIM + kk + cc * 8) * 2u);
            async_ld16(l, xb, g);
        }
        // B tiles: 4 experts x 64 rows x 32 bf16 = 1024 chunks, 4 per thread
#pragma unroll
        for (int c = 0; c < 4; ++c) {
            const int idx = tid + c * 256;
            const int e   = idx >> 8;
            const int rem = idx & 255;
            const int row = rem >> 2;
            const int cc  = rem & 3;
            const unsigned l = lds_off(&sB[b][(e * 64 + row) * 32 + cc * 8]);
            const unsigned g = (unsigned)(
                (((size_t)(e * OUTD + nBase + row)) * DIM + kk + cc * 8) * 2u);
            async_ld16(l, wb, g);
        }
    };

    v8f acc[NEXP][2][2];
    const v8f vzero = {0.f, 0.f, 0.f, 0.f, 0.f, 0.f, 0.f, 0.f};
#pragma unroll
    for (int e = 0; e < NEXP; ++e)
#pragma unroll
        for (int mi = 0; mi < 2; ++mi)
#pragma unroll
            for (int ni = 0; ni < 2; ++ni)
                acc[e][mi][ni] = vzero;

    const int r  = lane & 15;
    const int hi = lane >> 4;

    // prologue: fill buffer 0
    stage(0, 0);
    wait_async0();
    __syncthreads();

    for (int kk = 0; kk < DIM; kk += 32) {
        const int cur = (kk >> 5) & 1;

        // kick off DMA for the next K-step into the other buffer
        if (kk + 32 < DIM) stage(cur ^ 1, kk + 32);

        // ---- A fragments (16-bit A 16x32 layout: lane<16 K0-7/16-23,
        //      lane>=16 K8-15/24-31) ----
        v16bf afr[2];
#pragma unroll
        for (int mi = 0; mi < 2; ++mi) {
            const unsigned short* p = &sA[cur][(wm * 32 + mi * 16 + r) * 32];
            uint4 lo = *(const uint4*)(p + hi * 8);
            uint4 h2 = *(const uint4*)(p + 16 + hi * 8);
            FragU f;
            f.u[0] = lo.x; f.u[1] = lo.y; f.u[2] = lo.z; f.u[3] = lo.w;
            f.u[4] = h2.x; f.u[5] = h2.y; f.u[6] = h2.z; f.u[7] = h2.w;
            afr[mi] = f.v;
        }
        // ---- B fragments (16-bit B 32x16 layout: lane<16 K0-7/16-23,
        //      lane>=16 K8-15/24-31) + WMMA ----
#pragma unroll
        for (int e = 0; e < NEXP; ++e) {
#pragma unroll
            for (int ni = 0; ni < 2; ++ni) {
                const unsigned short* p =
                    &sB[cur][(e * 64 + wn * 32 + ni * 16 + r) * 32];
                uint4 lo = *(const uint4*)(p + hi * 8);
                uint4 h2 = *(const uint4*)(p + 16 + hi * 8);
                FragU f;
                f.u[0] = lo.x; f.u[1] = lo.y; f.u[2] = lo.z; f.u[3] = lo.w;
                f.u[4] = h2.x; f.u[5] = h2.y; f.u[6] = h2.z; f.u[7] = h2.w;
                const v16bf bfr = f.v;
#pragma unroll
                for (int mi = 0; mi < 2; ++mi) {
                    acc[e][mi][ni] = __builtin_amdgcn_wmma_f32_16x16x32_bf16(
                        false, afr[mi], false, bfr, (short)0, acc[e][mi][ni],
                        false, false);
                }
            }
        }

        // DMA for next buffer complete + all waves done reading current
        wait_async0();
        __syncthreads();
    }

    // ---- epilogue: gate-weighted combine (fp32) and store ----
    // C layout: VGPR i -> row = i + 8*(lane>=16); col = lane&15
#pragma unroll
    for (int mi = 0; mi < 2; ++mi) {
#pragma unroll
        for (int i = 0; i < 8; ++i) {
            const int row = mBase + wm * 32 + mi * 16 + hi * 8 + i;
            const float4 g = *(const float4*)(gate + (size_t)row * 4);
#pragma unroll
            for (int ni = 0; ni < 2; ++ni) {
                const int col = nBase + wn * 32 + ni * 16 + r;
                const float v = g.x * acc[0][mi][ni][i] + g.y * acc[1][mi][ni][i]
                              + g.z * acc[2][mi][ni][i] + g.w * acc[3][mi][ni][i];
                out[(size_t)row * OUTD + col] = v;
            }
        }
    }
}

// ---------------------------------------------------------------------------
extern "C" void kernel_launch(void* const* d_in, const int* in_sizes, int n_in,
                              void* d_out, int out_size, void* d_ws, size_t ws_size,
                              hipStream_t stream) {
    const float* x  = (const float*)d_in[0];  // [4,4096,1024]
    const float* We = (const float*)d_in[1];  // [4,1024,1024]
    const float* gw = (const float*)d_in[2];  // [4,1024]
    const float* gb = (const float*)d_in[3];  // [4]
    float* out = (float*)d_out;               // [4,4096,1024]

    // workspace layout (bytes):
    //   xb   bf16 [MTOT, DIM]        : 32 MB
    //   wb   bf16 [NEXP, OUTD, DIM]  :  8 MB
    //   gate f32  [MTOT, NEXP]       : 256 KB
    char* ws = (char*)d_ws;
    unsigned short* xbp = (unsigned short*)ws;
    size_t off = (size_t)MTOT * DIM * sizeof(unsigned short);
    unsigned short* wbp = (unsigned short*)(ws + off);
    off += (size_t)NEXP * OUTD * DIM * sizeof(unsigned short);
    float* gatep = (float*)(ws + off);

    gate_and_cvt_x<<<MTOT / 8, 256, 0, stream>>>(x, gw, gb, gatep, xbp);
    cvt_w<<<(NEXP * OUTD * DIM) / 1024, 256, 0, stream>>>(We, wbp);
    dim3 grid(OUTD / 64, MTOT / 128);
    moe_gemm<<<grid, 256, 0, stream>>>(xbp, wbp, gatep, out);
}
